// WignerCombiningSingleUnrolled_51702816309475
// MI455X (gfx1250) — compile-verified, matrix operand order checked by reading, not run on hardware
//
#include <hip/hip_runtime.h>
#include <hip/hip_bf16.h>

// ---------------------------------------------------------------------------
// out[n, s] = sum_t mult[t] * X1f[n, m1*9+m1p] * X2f[n, m2*9+m2p],  s = seg[t]
// GEMM form: out = P @ W^T,  K padded as k = a*96 + b  (a,b in [0,81))
//   P[n, k] = X1f[n, a] * X2f[n, b]   (synthesized in registers per tile)
//   W[s, k] = mult scattered into bf16 WMMA B-fragment layout (1.49 MB, L2)
// Block = 64 batches x 96 cols: 24 waves, each owns one 16x16 C tile and
// marches K in 32-chunks with v_wmma_f32_16x16x32_bf16. X1/X2 tiles staged
// into LDS by the Tensor Data Mover (tensor_load_to_lds + s_wait_tensorcnt).
// B fragments are software-pipelined one chunk ahead; A fragments cost
// 8x v_pk_mul_f32 + 8x v_cvt_pk_bf16_f32 per WMMA.
// ---------------------------------------------------------------------------

typedef __attribute__((ext_vector_type(16))) __bf16    v16bf;
typedef __attribute__((ext_vector_type(8)))  float     v8f;
typedef __attribute__((ext_vector_type(4)))  unsigned  v4u;
typedef __attribute__((ext_vector_type(8)))  int       v8i;
typedef __attribute__((ext_vector_type(4)))  int       v4i;

#define LDIM        81                       // 9*9 row length of X1/X2/out
#define BPAD        96                       // b padded to 3 chunks of 32
#define KTOT        (LDIM * BPAD)            // 7776
#define KCHUNKS     (KTOT / 32)              // 243
#define COLTILES    6                        // ceil(81/16)
#define MTILES      4
#define BATCH_TILE  (MTILES * 16)            // 64 batches per block
#define THREADS     (COLTILES * MTILES * 32) // 768 threads = 24 waves
#define WFRAG_HALVES (COLTILES * KCHUNKS * 32 * 16)  // 746,496 bf16 = 1.49 MB
// one extra zeroed chunk so the B-pipeline may over-read past the end
#define WFRAG_HALVES_PAD (WFRAG_HALVES + 32 * 16)

#if __has_builtin(__builtin_amdgcn_tensor_load_to_lds) && \
    __has_builtin(__builtin_amdgcn_s_wait_tensorcnt)
#define HAVE_TDM 1
#endif

__device__ __forceinline__ unsigned short f32_to_bf16_rne(float f) {
  unsigned u = __builtin_bit_cast(unsigned, f);
  return (unsigned short)((u + 0x7FFFu + ((u >> 16) & 1u)) >> 16);
}

// -------- prep 1: zero the dense-W fragment buffer in d_ws -------------------
__global__ void zero_ws_kernel(unsigned* __restrict__ w, int nwords) {
  int i = blockIdx.x * blockDim.x + threadIdx.x;
  if (i < nwords) w[i] = 0u;
}

// -------- prep 2: scatter sparse terms into WMMA B-fragment layout -----------
// B tile (32K x 16N bf16) per-lane layout (ISA 16-bit fragment layout):
//   lanes 0-15  (N = lane)      : slots 0-7 -> K 0-7,  slots 8-15 -> K 16-23
//   lanes 16-31 (N = lane - 16) : slots 0-7 -> K 8-15, slots 8-15 -> K 24-31
__global__ void build_wfrag_kernel(const float* __restrict__ mult,
                                   const int* __restrict__ m1,  const int* __restrict__ m1p,
                                   const int* __restrict__ m2,  const int* __restrict__ m2p,
                                   const int* __restrict__ seg,
                                   unsigned short* __restrict__ wfrag, int T) {
  int t = blockIdx.x * blockDim.x + threadIdx.x;
  if (t >= T) return;
  int s = seg[t];                                         // output column, 0..80
  int k = (m1[t] * 9 + m1p[t]) * BPAD + (m2[t] * 9 + m2p[t]);
  int colTile = s >> 4;
  int n       = s & 15;
  int ch      = k >> 5;
  int kin     = k & 31;
  int hi      = (kin >> 3) & 1;
  int lane    = n + (hi << 4);
  int slot    = (kin & 7) + ((kin & 16) ? 8 : 0);
  wfrag[((((long)colTile * KCHUNKS + ch) << 5) + lane) * 16 + slot] =
      f32_to_bf16_rne(mult[t]);
}

#ifdef HAVE_TDM
// Issue one TDM 2D tile load: rows x 81 f32, row stride 81, global -> LDS.
__device__ __forceinline__ void tdm_load_tile(unsigned lds_off, const float* gsrc,
                                              unsigned rows_valid) {
  unsigned long long ga = (unsigned long long)(const void*)gsrc;
  const unsigned dim0 = LDIM, tile0 = LDIM, tile1 = BATCH_TILE, stride0 = LDIM;
  v4u g0 = { 1u,                                     // count=1, user mode
             lds_off,                                // lds_addr
             (unsigned)ga,                           // global_addr[31:0]
             (unsigned)((ga >> 32) & 0x01FFFFFFull) | (2u << 30) };  // addr hi | type=2
  v8i g1 = { (int)(2u << 16),                                  // data_size = 4 bytes
             (int)((dim0 & 0xFFFFu) << 16),                    // tensor_dim0 lo16
             (int)((dim0 >> 16) | ((rows_valid & 0xFFFFu) << 16)), // dim0 hi / dim1 lo
             (int)((rows_valid >> 16) | ((tile0 & 0xFFFFu) << 16)),// dim1 hi / tile_dim0
             (int)(tile1 & 0xFFFFu),                           // tile_dim1 (tile_dim2=0)
             (int)stride0,                                     // tensor_dim0_stride lo32
             0, 0 };
  v4i gz = { 0, 0, 0, 0 };
#if __clang_major__ >= 23
  v8i gz8 = { 0, 0, 0, 0, 0, 0, 0, 0 };
  __builtin_amdgcn_tensor_load_to_lds(g0, g1, gz, gz, gz8, 0);
#else
  __builtin_amdgcn_tensor_load_to_lds(g0, g1, gz, gz, 0);
#endif
}
#endif

// -------- main: streaming WMMA GEMM -----------------------------------------
__global__ __launch_bounds__(THREADS)
void wigner_wmma_kernel(const float* __restrict__ X1, const float* __restrict__ X2,
                        const unsigned short* __restrict__ wfrag,
                        float* __restrict__ out, int nbatch) {
  // x2s first: x2s row-overflow reads (b in [81,96)) land in initialized LDS;
  // those lanes are multiplied by W == 0, so they contribute nothing.
  __shared__ float x2s[BATCH_TILE * LDIM];
  __shared__ float x1s[BATCH_TILE * LDIM];

  const int  tid      = threadIdx.x;
  const long base_row = (long)blockIdx.x * BATCH_TILE;
  const unsigned rows_valid =
      (unsigned)((nbatch - base_row) < BATCH_TILE ? (nbatch - base_row) : BATCH_TILE);

#ifdef HAVE_TDM
  if (tid < 32) {   // one wave drives the Tensor Data Mover for both tiles
    tdm_load_tile((unsigned)(unsigned long long)(void*)x1s, X1 + base_row * LDIM, rows_valid);
    tdm_load_tile((unsigned)(unsigned long long)(void*)x2s, X2 + base_row * LDIM, rows_valid);
    __builtin_amdgcn_s_wait_tensorcnt(0);
  }
#else
  {
    const long total = (long)nbatch * LDIM;
    for (int i = tid; i < BATCH_TILE * LDIM; i += THREADS) {
      long g = base_row * LDIM + i;
      float v1 = (g < total) ? X1[g] : 0.0f;
      float v2 = (g < total) ? X2[g] : 0.0f;
      x1s[i] = v1;
      x2s[i] = v2;
    }
  }
#endif
  __syncthreads();

  const int wave    = tid >> 5;           // 0..23
  const int lane    = tid & 31;
  const int colTile = wave % COLTILES;    // 0..5
  const int mtile   = wave / COLTILES;    // 0..3
  const int m       = (mtile << 4) + (lane & 15);  // batch row in tile, 0..63
  const int grp     = lane >> 4;          // K-half for A, M-half for C

  // Pre-load this lane's fixed X2 slice: the 48 values it feeds into A frags.
  float x2r[48];
#pragma unroll
  for (int cb = 0; cb < 3; ++cb) {
#pragma unroll
    for (int slot = 0; slot < 16; ++slot) {
      const int kin = slot + (grp << 3) + ((slot & 8) ? 8 : 0);
      x2r[cb * 16 + slot] = x2s[m * LDIM + cb * 32 + kin];
    }
  }

  v8f c = {0.f, 0.f, 0.f, 0.f, 0.f, 0.f, 0.f, 0.f};
  const v16bf* wp = (const v16bf*)wfrag + (long)colTile * KCHUNKS * 32 + lane;

  // Software pipeline: B fragment for chunk ch+1 is in flight while chunk ch's
  // WMMA executes. Last iteration over-reads one zero-padded chunk.
  v16bf bnext = wp[0];

  for (int a = 0; a < LDIM; ++a) {
    const float x1v = x1s[m * LDIM + a];                   // one scalar per 3 WMMAs
    __builtin_prefetch((const void*)(wp + (long)(a + 2) * 3 * 32), 0, 0);
#pragma unroll
    for (int cb = 0; cb < 3; ++cb) {
      v16bf bcur = bnext;
      bnext = wp[(long)(a * 3 + cb + 1) * 32];             // next chunk, 32 B/lane

      // 8x v_pk_mul_f32 + 8x v_cvt_pk_bf16_f32
      v16bf af;
#pragma unroll
      for (int slot = 0; slot < 16; ++slot) {
        af[slot] = (__bf16)(x1v * x2r[cb * 16 + slot]);
      }
      c = __builtin_amdgcn_wmma_f32_16x16x32_bf16(
          /*neg_a=*/false, af, /*neg_b=*/false, bcur,
          /*c_mod=*/(short)0, c, /*reuse_a=*/false, /*reuse_b=*/false);
    }
  }

  // C layout: lane -> N = lane&15; vgpr v -> M = v + 8*(lane>>4)
  const int col = (colTile << 4) + (lane & 15);
  if (col < LDIM) {
    const long row0 = base_row + (mtile << 4) + (grp << 3);
#pragma unroll
    for (int v = 0; v < 8; ++v) {
      const long row = row0 + v;
      if (row < nbatch) out[row * LDIM + col] = c[v];
    }
  }
}

// ---------------------------------------------------------------------------
extern "C" void kernel_launch(void* const* d_in, const int* in_sizes, int n_in,
                              void* d_out, int out_size, void* d_ws, size_t ws_size,
                              hipStream_t stream) {
  const float* X1   = (const float*)d_in[0];
  const float* X2   = (const float*)d_in[1];
  const float* mult = (const float*)d_in[2];
  const int*   m1   = (const int*)d_in[3];
  const int*   m1p  = (const int*)d_in[4];
  const int*   m2   = (const int*)d_in[5];
  const int*   m2p  = (const int*)d_in[6];
  const int*   seg  = (const int*)d_in[7];
  float*       out  = (float*)d_out;

  const int nbatch = in_sizes[0] / LDIM;    // X1 flat = N * 81
  const int T      = in_sizes[2];           // number of sparse terms

  unsigned short* wfrag = (unsigned short*)d_ws;   // needs ~1.5 MB of scratch
  const int nwords = WFRAG_HALVES_PAD / 2;

  zero_ws_kernel<<<(nwords + 255) / 256, 256, 0, stream>>>((unsigned*)d_ws, nwords);
  build_wfrag_kernel<<<(T + 255) / 256, 256, 0, stream>>>(mult, m1, m1p, m2, m2p,
                                                          seg, wfrag, T);
  const int nblocks = (nbatch + BATCH_TILE - 1) / BATCH_TILE;
  wigner_wmma_kernel<<<nblocks, THREADS, 0, stream>>>(X1, X2, wfrag, out, nbatch);
}
